// Decoder_34119220199812
// MI455X (gfx1250) — matrix operand order repaired and checked by reading, not compiled, not run on hardware
//
#include <hip/hip_runtime.h>
#include <hip/hip_bf16.h>
#include <math.h>

// ---------------------------------------------------------------------------
// Tacotron2 decoder, persistent-kernel implementation for gfx1250 (MI455X).
// 32 workgroups x 256 threads = 256 wave32s; one WMMA N-tile per wave.
// A-operands pre-packed to bf16 fragment layout and staged through LDS.
// ---------------------------------------------------------------------------

typedef __attribute__((ext_vector_type(16))) __bf16 v16bf;
typedef __attribute__((ext_vector_type(8)))  float  v8f;

#define NWG 32
#define NTH 256
#define NT  (NWG * NTH)          // 8192 threads, 256 waves

// Problem constants (from the reference)
#define Bx    32
#define TENC  400
#define EDIM  512
#define NMEL  80
#define PRE   256
#define ARNN  1024
#define DRNN  1024
#define ADIM  128
#define LF    32
#define LK    31
#define NGATE 4096
#define KATT  (PRE + EDIM + ARNN)   // 1792  (x = [p, ctx], then h)
#define KDEC  (ARNN + EDIM + DRNN)  // 2560  (x = [ah, ctx], then h)
#define NKA   (KATT / 32)           // 56 K-chunks
#define NKD   (KDEC / 32)           // 80 K-chunks
#define NTILE (NGATE / 16)          // 256 N-tiles == number of waves

// workspace float counts (state block must stay contiguous: zeroed by init)
#define STATE_FLOATS (4*Bx*1024 + 2*Bx*TENC + Bx*EDIM + Bx*NMEL)  // 175616

__device__ __forceinline__ float sigmf(float x) { return 1.0f / (1.0f + __expf(-x)); }

// Device-wide barrier: monotonically increasing counter, per-call epochs.
__device__ __forceinline__ void grid_barrier(unsigned* cnt, unsigned target) {
    __syncthreads();
    if (threadIdx.x == 0) {
        __threadfence();
        __hip_atomic_fetch_add(cnt, 1u, __ATOMIC_RELEASE, __HIP_MEMORY_SCOPE_AGENT);
        while (__hip_atomic_load(cnt, __ATOMIC_ACQUIRE, __HIP_MEMORY_SCOPE_AGENT) < target) {
            __builtin_amdgcn_s_sleep(1);
        }
    }
    __syncthreads();
}

// Copy one K-chunk of packed A (both 16-row tiles, 2KB) global -> LDS.
// 256 threads x 8 bytes each.
__device__ __forceinline__ void stage_A(const __bf16* __restrict__ XAP, int nK, int kc,
                                        __bf16* __restrict__ dst, int tid) {
    const int tile = tid >> 7;        // 0..1 (m-tile)
    const int e    = tid & 127;       // 4 halves each
    const unsigned long long* s =
        (const unsigned long long*)(XAP + (((size_t)tile * nK + kc) << 9) + e * 4);
    *(unsigned long long*)(dst + tile * 512 + e * 4) = *s;
}

// ---------------------------------------------------------------------------
// Gate GEMM:  GATES[32 x 4096] = X[32 x Ktot] @ Wcomb^T   (bf16 WMMA, f32 acc)
// Wp  : weights pre-packed in B-fragment order, contiguous k = 16*(lane>>4) + j
// XAP : activations pre-packed in A-fragment order, staged via LDS (lA[2][1024])
// One wave per 16-column tile; both 16-row tiles share each B fragment.
// ---------------------------------------------------------------------------
__device__ void gemm_gates(const __bf16* __restrict__ Wp, const __bf16* __restrict__ XAP,
                           float* __restrict__ GATES, int nK, __bf16* lA, int gtid) {
    const int tid  = threadIdx.x;
    const int wave = gtid >> 5;      // 0..255, == n-tile index
    const int lane = gtid & 31;
    const int nt   = wave;
    const int hi   = lane >> 4;

    v8f acc0 = {0.f, 0.f, 0.f, 0.f, 0.f, 0.f, 0.f, 0.f};
    v8f acc1 = acc0;

    stage_A(XAP, nK, 0, lA, tid);
    __syncthreads();

    for (int kc = 0; kc < nK; ++kc) {
        const int cur = kc & 1;
        if (kc + 1 < nK) stage_A(XAP, nK, kc + 1, lA + ((cur ^ 1) << 10), tid);

        const __bf16* bp = Wp + (((size_t)nt * nK + kc) << 9) + lane * 16;
        // WGP-scope prefetch of next B tile (weights are L2-resident; the
        // latency worth hiding is L2 -> WGP$, so use max temporal locality).
        if (kc + 1 < nK) __builtin_prefetch(bp + 512, 0, 3);   // global_prefetch_b8
        v16bf bfrag = *(const v16bf*)bp;                       // 2x global_load_b128

        const __bf16* la = lA + (cur << 10) + lane * 16;
        v16bf a0 = *(const v16bf*)la;                          // ds_load_b128
        v16bf a1 = *(const v16bf*)(la + 512);                  // ds_load_b128

        acc0 = __builtin_amdgcn_wmma_f32_16x16x32_bf16(false, a0, false, bfrag,
                                                       (short)0, acc0, false, false);
        acc1 = __builtin_amdgcn_wmma_f32_16x16x32_bf16(false, a1, false, bfrag,
                                                       (short)0, acc1, false, false);
        __syncthreads();
    }

    // C/D layout: lanes 0-15 -> N=lane, M=r ; lanes 16-31 -> N=lane-16, M=r+8
    const int nl = lane & 15;
    const int m0 = hi << 3;
    float* g0 = GATES + nt * 16 + nl;
#pragma unroll
    for (int r = 0; r < 8; ++r) {
        g0[(size_t)(m0 + r) * NGATE]      = acc0[r];
        g0[(size_t)(16 + m0 + r) * NGATE] = acc1[r];
    }
}

// Pack row-major f32 activations X[32 x Ktot] into bf16 A-fragment order:
//   idx = (mt*nK + kc)*512 + lane*16 + j
//   m = mt*16 + (lane&15) ; hi = lane>>4 ; v=j>>1 ; h=j&1
//   k = kc*32 + (hi<<4) + j
__device__ __forceinline__ void pack_A(const float* __restrict__ X, __bf16* __restrict__ XAP,
                                       int nK, int Ktot, int gtid) {
    const int total = 2 * nK * 512;
    for (int idx = gtid; idx < total; idx += NT) {
        const int mt = idx / (nK * 512);
        const int r2 = idx - mt * nK * 512;
        const int kc = r2 >> 9, r = r2 & 511;
        const int lane = r >> 4, j = r & 15;
        const int hi = lane >> 4, ml = lane & 15;
        const int v = j >> 1, h = j & 1;
        const int kl = (hi << 4) + j;
        XAP[idx] = (__bf16)X[(size_t)(mt * 16 + ml) * Ktot + kc * 32 + kl];
    }
}

__device__ void lstm_pointwise(const float* __restrict__ GATES,
                               const float* __restrict__ bih, const float* __restrict__ bhh,
                               float* __restrict__ H, float* __restrict__ C, int gtid) {
    for (int idx = gtid; idx < Bx * 1024; idx += NT) {
        const int b = idx >> 10, n = idx & 1023;
        const float* g = GATES + (size_t)b * NGATE;
        const float gi = g[n]        + bih[n]        + bhh[n];
        const float gf = g[1024 + n] + bih[1024 + n] + bhh[1024 + n];
        const float gg = g[2048 + n] + bih[2048 + n] + bhh[2048 + n];
        const float go = g[3072 + n] + bih[3072 + n] + bhh[3072 + n];
        const float c  = sigmf(gf) * C[idx] + sigmf(gi) * tanhf(gg);
        C[idx] = c;
        H[idx] = sigmf(go) * tanhf(c);
    }
}

__global__ void __launch_bounds__(NTH, 1)
tacotron_decoder(const float* __restrict__ memory, const int* __restrict__ n_steps_p,
                 const float* __restrict__ pw1,  const float* __restrict__ pw2,
                 const float* __restrict__ awih, const float* __restrict__ awhh,
                 const float* __restrict__ abih, const float* __restrict__ abhh,
                 const float* __restrict__ dwih, const float* __restrict__ dwhh,
                 const float* __restrict__ dbih, const float* __restrict__ dbhh,
                 const float* __restrict__ qw,   const float* __restrict__ mw,
                 const float* __restrict__ vw,   const float* __restrict__ cw,
                 const float* __restrict__ ldw,  const float* __restrict__ projw,
                 const float* __restrict__ projb,const float* __restrict__ gw,
                 const float* __restrict__ gb,
                 float* __restrict__ out, unsigned char* __restrict__ ws) {
    const int gtid = blockIdx.x * NTH + threadIdx.x;
    const int S = *n_steps_p;

    __shared__ __bf16 lA[2 * 1024];   // double-buffered A fragments (4KB)

    // ---- workspace layout (state block first: zeroed by init kernel) ----
    unsigned* cnt = (unsigned*)ws;
    float* AH     = (float*)(ws + 256);
    float* AC     = AH  + Bx * 1024;
    float* DHs    = AC  + Bx * 1024;
    float* DCs    = DHs + Bx * 1024;
    float* AWv    = DCs + Bx * 1024;      // attention weights  [32][400]
    float* AWC    = AWv + Bx * TENC;      // cumulative         [32][400]
    float* CTX    = AWC + Bx * TENC;      // context            [32][512]
    float* DECIN  = CTX + Bx * EDIM;      // previous mel       [32][80]
    // scratch (written before read every call)
    float* P1     = DECIN + Bx * NMEL;    // prenet hidden      [32][256]
    float* XATT   = P1   + Bx * PRE;      // [32][1792]
    float* XDEC   = XATT + Bx * KATT;     // [32][2560]
    float* GATES  = XDEC + Bx * KDEC;     // [32][4096]
    float* PQ     = GATES + Bx * NGATE;   // [32][128]
    float* LOC    = PQ   + Bx * ADIM;     // [32][32][400]
    float* EN     = LOC  + Bx * LF * TENC;// [32][400]
    float* PM     = EN   + Bx * TENC;     // processed memory [32][400][128]
    __bf16* WA    = (__bf16*)(PM + (size_t)Bx * TENC * ADIM); // packed att W
    __bf16* WD    = WA + (size_t)NTILE * NKA * 512;           // packed dec W
    __bf16* XAPA  = WD + (size_t)NTILE * NKD * 512;           // packed att A [2*NKA*512]
    __bf16* XAPD  = XAPA + 2 * NKA * 512;                     // packed dec A [2*NKD*512]

    float* out_mel   = out;                               // [B][80][S]
    float* out_gate  = out + (size_t)Bx * NMEL * S;       // [B][S]
    float* out_align = out_gate + (size_t)Bx * S;         // [B][S][T]

    unsigned epoch = 0;
    auto bar = [&]() { grid_barrier(cnt, (++epoch) * NWG); };

    // ================= phase 0: pack weights + processed_memory ============
    // Pack combined [Ktot x 4096] W^T into B-fragment order:
    //   idx = tile*512 + lane*16 + j ; tile = nt*nK + kc
    //   k = kc*32 + (lane>=16)*16 + (j>>1)*2 + (j&1) ; n = nt*16 + (lane&15)
    for (int idx = gtid; idx < NTILE * NKA * 512; idx += NT) {
        const int t = idx >> 9, r = idx & 511;
        const int lane = r >> 4, j = r & 15;
        const int kc = t % NKA, ntl = t / NKA;
        const int k = kc * 32 + ((lane >> 4) << 4) + ((j >> 1) << 1) + (j & 1);
        const int n = ntl * 16 + (lane & 15);
        const float w = (k < PRE + EDIM) ? awih[(size_t)n * (PRE + EDIM) + k]
                                         : awhh[(size_t)n * ARNN + (k - (PRE + EDIM))];
        WA[idx] = (__bf16)w;
    }
    for (int idx = gtid; idx < NTILE * NKD * 512; idx += NT) {
        const int t = idx >> 9, r = idx & 511;
        const int lane = r >> 4, j = r & 15;
        const int kc = t % NKD, ntl = t / NKD;
        const int k = kc * 32 + ((lane >> 4) << 4) + ((j >> 1) << 1) + (j & 1);
        const int n = ntl * 16 + (lane & 15);
        const float w = (k < ARNN + EDIM) ? dwih[(size_t)n * (ARNN + EDIM) + k]
                                          : dwhh[(size_t)n * DRNN + (k - (ARNN + EDIM))];
        WD[idx] = (__bf16)w;
    }
    // processed_memory[b][t][a] = memory[b][t][:] . memory_w[a][:]
    for (int idx = gtid; idx < Bx * TENC * ADIM; idx += NT) {
        const int a = idx & 127, bt = idx >> 7;
        const float* mr = memory + (size_t)bt * EDIM;
        const float* wr = mw + (size_t)a * EDIM;
        float s = 0.f;
        for (int e = 0; e < EDIM; ++e) s += mr[e] * wr[e];
        PM[idx] = s;
    }
    bar();

    // ========================== 500-step scan ==============================
    for (int s = 0; s < S; ++s) {
        // --- stage A: prenet layer 1 + location conv (uses prev aw/awc) ---
        for (int idx = gtid; idx < Bx * PRE; idx += NT) {
            const int b = idx >> 8, n = idx & 255;
            const float* x = DECIN + b * NMEL;
            const float* wr = pw1 + (size_t)n * NMEL;
            float acc = 0.f;
            for (int k = 0; k < NMEL; ++k) acc += x[k] * wr[k];
            P1[idx] = fmaxf(acc, 0.f);
        }
        for (int idx = gtid; idx < Bx * LF * TENC; idx += NT) {
            const int t = idx % TENC, f = (idx / TENC) & 31, b = idx / (TENC * LF);
            const float* a0 = AWv + b * TENC;
            const float* a1 = AWC + b * TENC;
            const float* w0 = cw + (size_t)f * 2 * LK;
            const float* w1 = w0 + LK;
            const int k0 = (t < 15) ? (15 - t) : 0;
            const int k1 = (t > TENC - 17) ? (TENC + 15 - t) : LK;
            float acc = 0.f;
            for (int k = k0; k < k1; ++k) {
                const int tt = t + k - 15;
                acc += a0[tt] * w0[k] + a1[tt] * w1[k];
            }
            LOC[idx] = acc;   // layout [b][f][t]
        }
        bar();

        // --- stage B: prenet layer 2 -> x_att[:,0:256]; ctx, ah copies ---
        for (int idx = gtid; idx < Bx * PRE; idx += NT) {
            const int b = idx >> 8, n = idx & 255;
            const float* x = P1 + b * PRE;
            const float* wr = pw2 + (size_t)n * PRE;
            float acc = 0.f;
            for (int k = 0; k < PRE; ++k) acc += x[k] * wr[k];
            XATT[(size_t)b * KATT + n] = fmaxf(acc, 0.f);
        }
        for (int idx = gtid; idx < Bx * EDIM; idx += NT)
            XATT[(size_t)(idx >> 9) * KATT + PRE + (idx & 511)] = CTX[idx];
        for (int idx = gtid; idx < Bx * ARNN; idx += NT)
            XATT[(size_t)(idx >> 10) * KATT + PRE + EDIM + (idx & 1023)] = AH[idx];
        bar();

        // --- stage B2: pack x_att -> bf16 A fragments ---
        pack_A(XATT, XAPA, NKA, KATT, gtid);
        bar();

        // --- stage C/D: attention LSTM (WMMA GEMM + pointwise) ---
        gemm_gates(WA, XAPA, GATES, NKA, lA, gtid);
        bar();
        lstm_pointwise(GATES, abih, abhh, AH, AC, gtid);
        bar();

        // --- stage E: pq = ah @ query_w^T ---
        for (int idx = gtid; idx < Bx * ADIM; idx += NT) {
            const int b = idx >> 7, a = idx & 127;
            const float* h = AH + (size_t)b * ARNN;
            const float* wr = qw + (size_t)a * ARNN;
            float acc = 0.f;
            for (int k = 0; k < ARNN; ++k) acc += h[k] * wr[k];
            PQ[idx] = acc;
        }
        bar();

        // --- stage F: energies[b][t] = v . tanh(pq + loc@ld^T + pm) ---
        for (int idx = gtid; idx < Bx * TENC; idx += NT) {
            const int b = idx / TENC, t = idx % TENC;
            const float* pqb = PQ + b * ADIM;
            const float* pmr = PM + (size_t)idx * ADIM;
            const float* lb  = LOC + (size_t)b * LF * TENC + t;
            float e = 0.f;
            for (int a = 0; a < ADIM; ++a) {
                const float* lda = ldw + (size_t)a * LF;
                float lf = 0.f;
                for (int f = 0; f < LF; ++f) lf += lb[(size_t)f * TENC] * lda[f];
                e += vw[a] * tanhf(pqb[a] + lf + pmr[a]);
            }
            EN[idx] = e;
        }
        bar();

        // --- stage G: softmax rows (one wave per batch row) + awc + align out
        {
            const int wv = gtid >> 5, lane = gtid & 31;
            if (wv < Bx) {
                const int b = wv;
                float mx = -1e30f;
                for (int t = lane; t < TENC; t += 32) mx = fmaxf(mx, EN[b * TENC + t]);
                for (int off = 16; off; off >>= 1) mx = fmaxf(mx, __shfl_xor(mx, off, 32));
                float sum = 0.f;
                for (int t = lane; t < TENC; t += 32) {
                    const float e = __expf(EN[b * TENC + t] - mx);
                    AWv[b * TENC + t] = e;
                    sum += e;
                }
                for (int off = 16; off; off >>= 1) sum += __shfl_xor(sum, off, 32);
                const float inv = 1.0f / sum;
                float* al = out_align + (size_t)b * S * TENC + (size_t)s * TENC;
                for (int t = lane; t < TENC; t += 32) {
                    const float w = AWv[b * TENC + t] * inv;
                    AWv[b * TENC + t] = w;
                    AWC[b * TENC + t] += w;
                    al[t] = w;
                }
            }
        }
        bar();

        // --- stage H: context + stage copies into x_dec (ah, dh) ---
        for (int idx = gtid; idx < Bx * EDIM; idx += NT) {
            const int b = idx >> 9, e0 = idx & 511;
            const float* wrow = AWv + b * TENC;
            const float* mrow = memory + (size_t)b * TENC * EDIM + e0;
            float acc = 0.f;
            for (int t = 0; t < TENC; ++t) acc += wrow[t] * mrow[(size_t)t * EDIM];
            CTX[idx] = acc;
        }
        for (int idx = gtid; idx < Bx * ARNN; idx += NT)
            XDEC[(size_t)(idx >> 10) * KDEC + (idx & 1023)] = AH[idx];
        for (int idx = gtid; idx < Bx * DRNN; idx += NT)
            XDEC[(size_t)(idx >> 10) * KDEC + ARNN + EDIM + (idx & 1023)] = DHs[idx];
        bar();

        // --- stage I: copy new ctx into x_dec ---
        for (int idx = gtid; idx < Bx * EDIM; idx += NT)
            XDEC[(size_t)(idx >> 9) * KDEC + ARNN + (idx & 511)] = CTX[idx];
        bar();

        // --- stage I2: pack x_dec -> bf16 A fragments ---
        pack_A(XDEC, XAPD, NKD, KDEC, gtid);
        bar();

        // --- stage J/K: decoder LSTM (WMMA GEMM + pointwise) ---
        gemm_gates(WD, XAPD, GATES, NKD, lA, gtid);
        bar();
        lstm_pointwise(GATES, dbih, dbhh, DHs, DCs, gtid);
        bar();

        // --- stage L: projection + gate; mel feeds back as dec_in ---
        for (int idx = gtid; idx < Bx * NMEL + Bx; idx += NT) {
            if (idx < Bx * NMEL) {
                const int b = idx / NMEL, n = idx % NMEL;
                const float* wr = projw + (size_t)n * (DRNN + EDIM);
                const float* h = DHs + (size_t)b * DRNN;
                const float* c = CTX + (size_t)b * EDIM;
                float acc = projb[n];
                for (int k = 0; k < DRNN; ++k) acc += h[k] * wr[k];
                for (int k = 0; k < EDIM; ++k) acc += c[k] * wr[DRNN + k];
                out_mel[((size_t)b * NMEL + n) * S + s] = acc;
                DECIN[b * NMEL + n] = acc;
            } else {
                const int b = idx - Bx * NMEL;
                const float* h = DHs + (size_t)b * DRNN;
                const float* c = CTX + (size_t)b * EDIM;
                float acc = gb[0];
                for (int k = 0; k < DRNN; ++k) acc += h[k] * gw[k];
                for (int k = 0; k < EDIM; ++k) acc += c[k] * gw[DRNN + k];
                out_gate[(size_t)b * S + s] = acc;
            }
        }
        bar();   // protects dec_in / aw / awc / ctx for next iteration
    }
}

// Zero the barrier counter + recurrent state each call (ws is not re-poisoned
// between graph replays; decode must start from zeros deterministically).
__global__ void init_state(unsigned* __restrict__ w, int nwords) {
    for (int i = blockIdx.x * blockDim.x + threadIdx.x; i < nwords;
         i += gridDim.x * blockDim.x)
        w[i] = 0u;
}

extern "C" void kernel_launch(void* const* d_in, const int* in_sizes, int n_in,
                              void* d_out, int out_size, void* d_ws, size_t ws_size,
                              hipStream_t stream) {
    const float* memory  = (const float*)d_in[0];
    const int*   n_steps = (const int*)  d_in[1];
    const float* pw1     = (const float*)d_in[2];
    const float* pw2     = (const float*)d_in[3];
    const float* awih    = (const float*)d_in[4];
    const float* awhh    = (const float*)d_in[5];
    const float* abih    = (const float*)d_in[6];
    const float* abhh    = (const float*)d_in[7];
    const float* dwih    = (const float*)d_in[8];
    const float* dwhh    = (const float*)d_in[9];
    const float* dbih    = (const float*)d_in[10];
    const float* dbhh    = (const float*)d_in[11];
    const float* qw      = (const float*)d_in[12];
    const float* mw      = (const float*)d_in[13];
    const float* vw      = (const float*)d_in[14];
    const float* cw      = (const float*)d_in[15];
    const float* ldw     = (const float*)d_in[16];
    const float* projw   = (const float*)d_in[17];
    const float* projb   = (const float*)d_in[18];
    const float* gw      = (const float*)d_in[19];
    const float* gb      = (const float*)d_in[20];

    const int init_words = 64 + STATE_FLOATS;   // counter pad + state block
    init_state<<<256, 256, 0, stream>>>((unsigned*)d_ws, init_words);
    tacotron_decoder<<<NWG, NTH, 0, stream>>>(
        memory, n_steps, pw1, pw2, awih, awhh, abih, abhh, dwih, dwhh,
        dbih, dbhh, qw, mw, vw, cw, ldw, projw, projb, gw, gb,
        (float*)d_out, (unsigned char*)d_ws);
}
